// DistangledMultiHeadAttention_7533372637668
// MI455X (gfx1250) — matrix operand order repaired and checked by reading, compiled, not run on hardware
//
#include <hip/hip_runtime.h>
#include <hip/hip_bf16.h>

// ---------------------------------------------------------------------------
// DistangledMultiHeadAttention for MI455X (gfx1250, wave32, WMMA bf16)
// B=8, N=1024, HID=1024, H=16, D=64
// ---------------------------------------------------------------------------

typedef __bf16 bf16;
typedef __attribute__((ext_vector_type(16))) __bf16 v16bf;
typedef __attribute__((ext_vector_type(8)))  __bf16 v8bf;
typedef __attribute__((ext_vector_type(8)))  float  v8f;

#define Bdim   8
#define Ndim   1024
#define HIDdim 1024
#define Hdim   16
#define Ddim   64

__device__ inline v8f wmma_bf16(v16bf a, v16bf b, v8f c) {
    // D = A(16x32) * B(32x16) + C, f32 accumulate
    return __builtin_amdgcn_wmma_f32_16x16x32_bf16(
        /*neg_a=*/false, a, /*neg_b=*/false, b,
        /*c_mod=*/(short)0, c, /*reuse_a=*/false, /*reuse_b=*/false);
}

union ABFrag { v16bf v; v8bf h[2]; };

// A fragment: 16x32 bf16 tile, element (m, k) at base[(row0+m)*ld + k0 + k].
// ISA layout: lane<16 holds row=lane, K {0..7, 16..23}; lane>=16 holds K {8..15, 24..31}.
__device__ inline v16bf load_afrag(const bf16* base, int ld, int row0, int k0, int lane) {
    int hi = lane >> 4, mr = lane & 15;
    const bf16* p = base + (size_t)(row0 + mr) * ld + k0 + hi * 8;
    ABFrag f;
    f.h[0] = *(const v8bf*)(p);
    f.h[1] = *(const v8bf*)(p + 16);
    return f.v;
}

// B fragment: 32x16 bf16 tile, element (k, n) at base[(col0+n)*ld + k0 + k]
// (i.e. K contiguous per column). Lane<16: col=lane, K {0..7, 16..23}; lane>=16: K {8..15, 24..31}.
__device__ inline v16bf load_bfrag(const bf16* base, int ld, int col0, int k0, int lane) {
    int hi = lane >> 4, cn = lane & 15;
    const bf16* p = base + (size_t)(col0 + cn) * ld + k0 + hi * 8;
    ABFrag f;
    f.h[0] = *(const v8bf*)(p);
    f.h[1] = *(const v8bf*)(p + 16);
    return f.v;
}

// ---------------------------------------------------------------------------
// 1) f32 -> bf16 convert (row-major preserved)
// ---------------------------------------------------------------------------
__global__ void __launch_bounds__(256)
cvt_bf16_kernel(const float* __restrict__ src, bf16* __restrict__ dst, int n) {
    int i = (blockIdx.x * 256 + threadIdx.x) * 4;
    if (i < n) {
        float4 v = *(const float4*)(src + i);
        dst[i + 0] = (bf16)v.x;
        dst[i + 1] = (bf16)v.y;
        dst[i + 2] = (bf16)v.z;
        dst[i + 3] = (bf16)v.w;
    }
}

// ---------------------------------------------------------------------------
// 2) W[K,Nout] f32 -> WT[Nout,K] bf16 (LDS tiled transpose)
// ---------------------------------------------------------------------------
__global__ void __launch_bounds__(256)
transpose_cvt_kernel(const float* __restrict__ W, bf16* __restrict__ WT,
                     int K, int Nout) {
    __shared__ float tile[32][33];
    int tx = threadIdx.x & 31, ty = threadIdx.x >> 5;   // 32 x 8
    int kb = blockIdx.y * 32, nb = blockIdx.x * 32;
#pragma unroll
    for (int i = 0; i < 4; i++) {
        int k = kb + ty + i * 8;
        tile[ty + i * 8][tx] = W[(size_t)k * Nout + nb + tx];
    }
    __syncthreads();
#pragma unroll
    for (int i = 0; i < 4; i++) {
        int n = nb + ty + i * 8;
        WT[(size_t)n * K + kb + tx] = (bf16)tile[tx][ty + i * 8];
    }
}

// ---------------------------------------------------------------------------
// 3) WMMA GEMM: C[M=8192, Nout=1024] = A_bf16[8192,1024] @ WT_bf16[1024,1024]^T + bias
// Block: 256 thr (8 waves, 4x2), block tile 128x64, wave tile 32x32.
// mode 0: f32 -> Qp[B,H,N,D]   mode 1: f32 -> Kp[B,H,N,D]
// mode 2: bf16 -> Vt[B,H,D,N]  mode 3: f32 -> out[M,1024] (row major)
// ---------------------------------------------------------------------------
__global__ void __launch_bounds__(256)
gemm_wmma_kernel(const bf16* __restrict__ A, const bf16* __restrict__ WT,
                 const float* __restrict__ bias, void* __restrict__ outp, int mode) {
    const int tid = threadIdx.x;
    const int wave = tid >> 5, lane = tid & 31;
    const int mwave = wave & 3, nwave = wave >> 2;
    const int Mbase = blockIdx.y * 128 + mwave * 32;
    const int Nbase = blockIdx.x * 64 + nwave * 32;

    const v8f z = {0.f, 0.f, 0.f, 0.f, 0.f, 0.f, 0.f, 0.f};
    v8f acc[2][2] = {{z, z}, {z, z}};

    for (int kk = 0; kk < 1024; kk += 32) {
        v16bf a0 = load_afrag(A, 1024, Mbase,      kk, lane);
        v16bf a1 = load_afrag(A, 1024, Mbase + 16, kk, lane);
        v16bf b0 = load_bfrag(WT, 1024, Nbase,      kk, lane);
        v16bf b1 = load_bfrag(WT, 1024, Nbase + 16, kk, lane);
        acc[0][0] = wmma_bf16(a0, b0, acc[0][0]);
        acc[0][1] = wmma_bf16(a0, b1, acc[0][1]);
        acc[1][0] = wmma_bf16(a1, b0, acc[1][0]);
        acc[1][1] = wmma_bf16(a1, b1, acc[1][1]);
    }

    const int hi = lane >> 4, ln = lane & 15;
#pragma unroll
    for (int Mi = 0; Mi < 2; Mi++)
#pragma unroll
        for (int Ni = 0; Ni < 2; Ni++)
#pragma unroll
            for (int r = 0; r < 8; r++) {
                int m    = Mbase + Mi * 16 + r + hi * 8;   // global row (b*N + n)
                int nout = Nbase + Ni * 16 + ln;           // output column
                float val = acc[Mi][Ni][r] + bias[nout];
                int b = m >> 10, n = m & 1023;
                int h = nout >> 6, d = nout & 63;
                if (mode == 0 || mode == 1) {
                    float* o = (float*)outp;
                    o[(((size_t)(b * Hdim + h)) * Ndim + n) * Ddim + d] = val;
                } else if (mode == 2) {
                    bf16* o = (bf16*)outp;
                    o[(((size_t)(b * Hdim + h)) * Ddim + d) * Ndim + n] = (bf16)val;
                } else {
                    float* o = (float*)outp;
                    o[(size_t)m * 1024 + nout] = val;
                }
            }
}

// ---------------------------------------------------------------------------
// 4) unary: u[b,h,:] = softmax_n( k[b,n,:] @ Wu[:,h] + bu[h] )
// One block per (b,h).
// ---------------------------------------------------------------------------
__global__ void __launch_bounds__(256)
unary_softmax_kernel(const float* __restrict__ kin, const float* __restrict__ Wu,
                     const float* __restrict__ bu, float* __restrict__ uout) {
    __shared__ float wu[1024];
    __shared__ float vals[1024];
    __shared__ float red[256];
    const int bh = blockIdx.x, b = bh >> 4, h = bh & 15, t = threadIdx.x;

    for (int i = t; i < 1024; i += 256) wu[i] = Wu[(size_t)i * Hdim + h];
    __syncthreads();

    const float bias = bu[h];
    for (int j = 0; j < 4; j++) {
        int n = t + j * 256;
        const float* krow = kin + ((size_t)b * Ndim + n) * HIDdim;
        float acc = 0.f;
        for (int c = 0; c < 1024; c += 4) {
            float4 kv = *(const float4*)(krow + c);
            acc += kv.x * wu[c] + kv.y * wu[c + 1] + kv.z * wu[c + 2] + kv.w * wu[c + 3];
        }
        vals[n] = acc + bias;
    }
    __syncthreads();

    float lm = fmaxf(fmaxf(vals[t], vals[t + 256]), fmaxf(vals[t + 512], vals[t + 768]));
    red[t] = lm; __syncthreads();
    for (int s = 128; s > 0; s >>= 1) { if (t < s) red[t] = fmaxf(red[t], red[t + s]); __syncthreads(); }
    float mx = red[0]; __syncthreads();

    float ls = 0.f;
    for (int j = 0; j < 4; j++) { int n = t + j * 256; float e = __expf(vals[n] - mx); vals[n] = e; ls += e; }
    red[t] = ls; __syncthreads();
    for (int s = 128; s > 0; s >>= 1) { if (t < s) red[t] += red[t + s]; __syncthreads(); }
    float inv = 1.0f / red[0];
    for (int j = 0; j < 4; j++) { int n = t + j * 256; uout[(size_t)bh * Ndim + n] = vals[n] * inv; }
}

// ---------------------------------------------------------------------------
// 5) mean-center over N for Q (scale by D^-0.5) and K, emit bf16 [B,H,N,D]
// One block per (b,h).
// ---------------------------------------------------------------------------
__global__ void __launch_bounds__(256)
mean_center_kernel(const float* __restrict__ Qp, const float* __restrict__ Kp,
                   bf16* __restrict__ qc, bf16* __restrict__ kc) {
    __shared__ float red[4][64];
    __shared__ float meanv[64];
    const int bh = blockIdx.x, t = threadIdx.x;

    for (int pass = 0; pass < 2; ++pass) {
        const float* s = (pass ? Kp : Qp) + (size_t)bh * Ndim * Ddim;
        bf16* dst = (pass ? kc : qc) + (size_t)bh * Ndim * Ddim;
        const float mul = pass ? 1.0f : 0.125f;   // 64^-0.5 = 1/8

        int dd = t & 63, c = t >> 6;
        float sum = 0.f;
        for (int i = 0; i < 256; i++) sum += s[(size_t)(c * 256 + i) * Ddim + dd];
        red[c][dd] = sum;
        __syncthreads();
        if (t < 64)
            meanv[t] = (red[0][t] + red[1][t] + red[2][t] + red[3][t]) * (1.0f / 1024.0f);
        __syncthreads();

        for (int base = 0; base < Ndim * Ddim; base += 256) {
            int idx = base + t;
            dst[idx] = (bf16)((s[idx] - meanv[idx & 63]) * mul);
        }
        __syncthreads();
    }
}

// ---------------------------------------------------------------------------
// 6) attention: flash-style online softmax + separate unnormalized (adj*u)@V path
// Block = 128 thr (4 waves), wave owns 16 rows, grid = B*H*(N/64).
// out = (adj .* u) @ V + softmax(mask(S)) @ V          (uadj=1)
// out =           u @ V + softmax(S) @ V               (uadj=0)
// Writes bf16 attn_out[B,N,H*D] (row-major, A-operand of output projection).
// ---------------------------------------------------------------------------
__global__ void __launch_bounds__(128)
attn_kernel(const bf16* __restrict__ qc, const bf16* __restrict__ kc,
            const bf16* __restrict__ Vt, const float* __restrict__ u,
            const int* __restrict__ adj, const int* __restrict__ use_adj_p,
            bf16* __restrict__ attn_out) {
    __shared__ bf16 ldsP[4][16 * 32];
    __shared__ bf16 ldsPU[4][16 * 32];

    const int tid = threadIdx.x, wave = tid >> 5, lane = tid & 31;
    const int hi = lane >> 4, ln = lane & 15;
    const int bh = blockIdx.x >> 4;          // b*H + h
    const int rt = blockIdx.x & 15;          // 64-row tile index
    const int b = bh >> 4, h = bh & 15;
    const int rowbase = rt * 64 + wave * 16;
    const int uadj = use_adj_p[0];

    const bf16*  qcb = qc + (size_t)bh * Ndim * Ddim;
    const bf16*  kcb = kc + (size_t)bh * Ndim * Ddim;
    const bf16*  vtb = Vt + (size_t)bh * Ddim * Ndim;
    const float* ub  = u  + (size_t)bh * Ndim;
    const int*   adjb = adj + (size_t)b * Ndim * Ndim;

    // Q fragments for this wave's 16 rows (K = D = 64 -> 2 frags)
    const v16bf qa0 = load_afrag(qcb, Ddim, rowbase, 0,  lane);
    const v16bf qa1 = load_afrag(qcb, Ddim, rowbase, 32, lane);

    const v8f z = {0.f, 0.f, 0.f, 0.f, 0.f, 0.f, 0.f, 0.f};
    v8f accS[4] = {z, z, z, z};   // softmax path (rescaled online, /= l at end)
    v8f accU[4] = {z, z, z, z};   // (adj*u) path (no normalization)
    float m[8], l[8];
#pragma unroll
    for (int r = 0; r < 8; r++) { m[r] = -1e30f; l[r] = 0.f; }

    bf16* P  = ldsP[wave];
    bf16* PU = ldsPU[wave];

    for (int j = 0; j < Ndim; j += 32) {
        // ---- scores: two 16x16 tiles over cols [j, j+32)
        v8f s0 = z, s1 = z;
        s0 = wmma_bf16(qa0, load_bfrag(kcb, Ddim, j,      0,  lane), s0);
        s0 = wmma_bf16(qa1, load_bfrag(kcb, Ddim, j,      32, lane), s0);
        s1 = wmma_bf16(qa0, load_bfrag(kcb, Ddim, j + 16, 0,  lane), s1);
        s1 = wmma_bf16(qa1, load_bfrag(kcb, Ddim, j + 16, 32, lane), s1);

        const int c0 = j + ln, c1 = j + 16 + ln;
        const float u0 = ub[c0], u1 = ub[c1];

        float e0[8], e1[8], pu0[8], pu1[8], scale[8];
#pragma unroll
        for (int r = 0; r < 8; r++) {
            int row = rowbase + r + hi * 8;
            float a0 = 1.f, a1 = 1.f;
            if (uadj) {
                a0 = (float)adjb[(size_t)row * Ndim + c0];
                a1 = (float)adjb[(size_t)row * Ndim + c1];
            }
            bool msk0 = uadj && (a0 == 0.f);
            bool msk1 = uadj && (a1 == 0.f);
            float sv0 = msk0 ? -1e30f : s0[r];
            float sv1 = msk1 ? -1e30f : s1[r];
            pu0[r] = a0 * u0;
            pu1[r] = a1 * u1;

            // row max over 32 cols: local, then across the 16 lanes of this half-wave
            float tm = fmaxf(sv0, sv1);
            for (int off = 1; off < 16; off <<= 1)
                tm = fmaxf(tm, __shfl_xor(tm, off, 16));
            float mn = fmaxf(m[r], tm);
            scale[r] = __expf(m[r] - mn);
            e0[r] = msk0 ? 0.f : __expf(sv0 - mn);
            e1[r] = msk1 ? 0.f : __expf(sv1 - mn);
            float ts = e0[r] + e1[r];
            for (int off = 1; off < 16; off <<= 1)
                ts += __shfl_xor(ts, off, 16);
            l[r] = l[r] * scale[r] + ts;
            m[r] = mn;
        }

        // rescale softmax accumulator
#pragma unroll
        for (int t = 0; t < 4; t++)
#pragma unroll
            for (int r = 0; r < 8; r++) accS[t][r] *= scale[r];

        // C-layout -> A-layout bounce through per-wave LDS (row-major 16x32)
#pragma unroll
        for (int r = 0; r < 8; r++) {
            int row = r + hi * 8;
            P [row * 32 + ln]      = (bf16)e0[r];
            P [row * 32 + 16 + ln] = (bf16)e1[r];
            PU[row * 32 + ln]      = (bf16)pu0[r];
            PU[row * 32 + 16 + ln] = (bf16)pu1[r];
        }
        v16bf pf  = load_afrag(P,  32, 0, 0, lane);
        v16bf puf = load_afrag(PU, 32, 0, 0, lane);

        // P(16x32) @ V(32x64): B frags contiguous along seq in Vt[B,H,D,N]
#pragma unroll
        for (int t = 0; t < 4; t++) {
            v16bf vb = load_bfrag(vtb, Ndim, t * 16, j, lane);
            accS[t] = wmma_bf16(pf,  vb, accS[t]);
            accU[t] = wmma_bf16(puf, vb, accU[t]);
        }
    }

    // finalize: out = accU + accS / l, store bf16 [B,N,H*D]
    float inv[8];
#pragma unroll
    for (int r = 0; r < 8; r++) inv[r] = 1.0f / l[r];
#pragma unroll
    for (int t = 0; t < 4; t++)
#pragma unroll
        for (int r = 0; r < 8; r++) {
            int row = rowbase + r + hi * 8;
            int d = t * 16 + ln;
            float o = accU[t][r] + accS[t][r] * inv[r];
            attn_out[((size_t)b * Ndim + row) * HIDdim + h * Ddim + d] = (bf16)o;
        }
}

// ---------------------------------------------------------------------------
// host launcher
// ---------------------------------------------------------------------------
extern "C" void kernel_launch(void* const* d_in, const int* in_sizes, int n_in,
                              void* d_out, int out_size, void* d_ws, size_t ws_size,
                              hipStream_t stream) {
    const float* q   = (const float*)d_in[0];
    const float* k   = (const float*)d_in[1];
    const float* vin = (const float*)d_in[2];
    const int*   adj = (const int*)d_in[3];
    const int*   use_adj = (const int*)d_in[4];
    const float* Wq = (const float*)d_in[5];
    const float* bq = (const float*)d_in[6];
    const float* Wk = (const float*)d_in[7];
    const float* bk = (const float*)d_in[8];
    const float* Wv = (const float*)d_in[9];
    const float* bv = (const float*)d_in[10];
    const float* Wu = (const float*)d_in[11];
    const float* bu = (const float*)d_in[12];
    const float* Wo = (const float*)d_in[13];
    const float* bo = (const float*)d_in[14];

    char* ws = (char*)d_ws;
    size_t off = 0;
    auto take = [&](size_t bytes) -> char* {
        char* p = ws + off;
        off = (off + bytes + 255) & ~(size_t)255;
        return p;
    };
    const size_t MT = (size_t)Bdim * Ndim;          // 8192 rows
    const size_t NT = MT * HIDdim;                  // 8M elements

    bf16* qb  = (bf16*)take(NT * 2);
    bf16* kb  = (bf16*)take(NT * 2);
    bf16* vb  = (bf16*)take(NT * 2);
    bf16* WqT = (bf16*)take((size_t)1024 * 1024 * 2);
    bf16* WkT = (bf16*)take((size_t)1024 * 1024 * 2);
    bf16* WvT = (bf16*)take((size_t)1024 * 1024 * 2);
    bf16* WoT = (bf16*)take((size_t)1024 * 1024 * 2);
    float* Qp = (float*)take(NT * 4);               // [B,H,N,D] f32
    float* Kp = (float*)take(NT * 4);               // [B,H,N,D] f32
    bf16* Vt  = (bf16*)take(NT * 2);                // [B,H,D,N] bf16
    bf16* qcc = (bf16*)take(NT * 2);                // centered Q bf16 [B,H,N,D]
    bf16* kcc = (bf16*)take(NT * 2);                // centered K bf16 [B,H,N,D]
    float* uS = (float*)take((size_t)Bdim * Hdim * Ndim * 4);
    bf16* attn_bf = (bf16*)take(NT * 2);            // [B,N,H*D] bf16
    (void)ws_size; (void)in_sizes; (void)n_in; (void)out_size;

    // 1) activations -> bf16
    {
        dim3 g((unsigned)(NT / 1024));
        cvt_bf16_kernel<<<g, 256, 0, stream>>>(q,   qb, (int)NT);
        cvt_bf16_kernel<<<g, 256, 0, stream>>>(k,   kb, (int)NT);
        cvt_bf16_kernel<<<g, 256, 0, stream>>>(vin, vb, (int)NT);
    }
    // 2) weights -> bf16 transposed [out,in]
    {
        dim3 g(1024 / 32, 1024 / 32);
        transpose_cvt_kernel<<<g, 256, 0, stream>>>(Wq, WqT, 1024, 1024);
        transpose_cvt_kernel<<<g, 256, 0, stream>>>(Wk, WkT, 1024, 1024);
        transpose_cvt_kernel<<<g, 256, 0, stream>>>(Wv, WvT, 1024, 1024);
        transpose_cvt_kernel<<<g, 256, 0, stream>>>(Wo, WoT, 1024, 1024);
    }
    // 3) projections (WMMA)
    {
        dim3 g(1024 / 64, (unsigned)(MT / 128));
        gemm_wmma_kernel<<<g, 256, 0, stream>>>(qb, WqT, bq, (void*)Qp, 0);
        gemm_wmma_kernel<<<g, 256, 0, stream>>>(kb, WkT, bk, (void*)Kp, 1);
        gemm_wmma_kernel<<<g, 256, 0, stream>>>(vb, WvT, bv, (void*)Vt, 2);
    }
    // 4) unary softmax
    unary_softmax_kernel<<<Bdim * Hdim, 256, 0, stream>>>(k, Wu, bu, uS);
    // 5) mean-center Q/K -> bf16
    mean_center_kernel<<<Bdim * Hdim, 256, 0, stream>>>(Qp, Kp, qcc, kcc);
    // 6) attention (WMMA flash-style)
    attn_kernel<<<Bdim * Hdim * (Ndim / 64), 128, 0, stream>>>(
        qcc, kcc, Vt, uS, adj, use_adj, attn_bf);
    // 7) output projection (WMMA) -> d_out f32 [B,N,HID]
    {
        dim3 g(1024 / 64, (unsigned)(MT / 128));
        gemm_wmma_kernel<<<g, 256, 0, stream>>>(attn_bf, WoT, bo, d_out, 3);
    }
}
